// AttentionLayer_42150809043069
// MI455X (gfx1250) — matrix operand order repaired and checked by reading, not compiled, not run on hardware
//
#include <hip/hip_runtime.h>
#include <hip/hip_bf16.h>
#include <math.h>

#define B_   2
#define S_   2048
#define HID_ 2048
#define NH_  16
#define DH_  128
#define MTOT (B_ * S_)

typedef __bf16 bf16;
typedef __attribute__((ext_vector_type(16))) bf16  v16bf;
typedef __attribute__((ext_vector_type(8)))  bf16  bf16x8;
typedef __attribute__((ext_vector_type(8)))  float v8f;

__device__ __forceinline__ v8f wmma_bf16(v16bf a, v16bf b, v8f c) {
  // D = A(16x32 bf16) * B(32x16 bf16) + C(16x16 f32)
  return __builtin_amdgcn_wmma_f32_16x16x32_bf16(false, a, false, b, (short)0, c,
                                                 false, false);
}

// ---- Fragment gathers per CDNA5 ISA 7.12.2 VGPR layouts (wave32) ----
// A (16x32, 16-bit): lane l: m=l&15, h=l>>4; slot i: k = (i<8 ? 8h+i : 8+8h+i)
__device__ __forceinline__ v16bf load_afrag(const bf16* Sb, int row0, int ld,
                                            int kbase, int lane) {
  const int m = lane & 15, h = lane >> 4;
  const bf16* p = Sb + (row0 + m) * ld + kbase + 8 * h;
  bf16x8 r0 = *(const bf16x8*)(p);
  bf16x8 r1 = *(const bf16x8*)(p + 16);
  v16bf a;
#pragma unroll
  for (int i = 0; i < 8; i++) { a[i] = r0[i]; a[i + 8] = r1[i]; }
  return a;
}
// B (32x16, 16-bit): lane l: n=l&15, h=l>>4; slot i: k = (i<8 ? 8h+i : 8+8h+i)
// LDS tile stored as [N][K] row-major (i.e. already transposed vs math layout)
__device__ __forceinline__ v16bf load_bfrag(const bf16* Sb, int row0, int ld,
                                            int kbase, int lane) {
  const int n = lane & 15, h = lane >> 4;
  const bf16* p = Sb + (row0 + n) * ld + kbase + 8 * h;
  bf16x8 r0 = *(const bf16x8*)(p);
  bf16x8 r1 = *(const bf16x8*)(p + 16);
  v16bf b;
#pragma unroll
  for (int i = 0; i < 8; i++) { b[i] = r0[i]; b[i + 8] = r1[i]; }
  return b;
}

__device__ __forceinline__ void cvt_store16(bf16* dst, const float* src) {
  bf16x8 h0, h1;
#pragma unroll
  for (int i = 0; i < 8; i++) { h0[i] = (bf16)src[i]; h1[i] = (bf16)src[i + 8]; }
  *(bf16x8*)dst = h0;
  *(bf16x8*)(dst + 8) = h1;
}

// ================= Kernel 1: fused QKV projection (f32 in, bf16 out) =========
// 128x64 macro-tile, KTILE=64, double-buffered LDS, 8 waves:
// wave w owns rows 16w..16w+15 and all 64 N columns (4 WMMA tiles).
__global__ __launch_bounds__(256) void qkv_gemm_kernel(
    const float* __restrict__ X,
    const float* __restrict__ Wq, const float* __restrict__ Wk,
    const float* __restrict__ Wv,
    const float* __restrict__ bq, const float* __restrict__ bk,
    const float* __restrict__ bv,
    bf16* __restrict__ Qo, bf16* __restrict__ Ko, bf16* __restrict__ Vo) {
  __shared__ bf16 As[2][128 * 72];
  __shared__ bf16 Bs[2][64 * 72];
  const int t = threadIdx.x, lane = t & 31, w = t >> 5;
  const int bm = blockIdx.x, bn = blockIdx.y, z = blockIdx.z;
  const float* Wsel = (z == 0) ? Wq : (z == 1) ? Wk : Wv;
  const float* bsel = (z == 0) ? bq : (z == 1) ? bk : bv;
  bf16* Osel = (z == 0) ? Qo : (z == 1) ? Ko : Vo;

  v8f acc[4];
#pragma unroll
  for (int i = 0; i < 4; i++) acc[i] = (v8f){0.f, 0.f, 0.f, 0.f, 0.f, 0.f, 0.f, 0.f};

  // staging: A 128x64 f32 -> 32 floats/thread; B 64x64 f32 -> 16 floats/thread
  const int ar = t >> 1, ac = (t & 1) * 32;
  const int br = t >> 2, bc = (t & 3) * 16;
  const float* xrow = X + (size_t)(bm * 128 + ar) * HID_ + ac;
  const float* wrow = Wsel + (size_t)(bn * 64 + br) * HID_ + bc;
  const int aoff = ar * 72 + ac;
  const int boff = br * 72 + bc;

  float fx[32], fw[16];
#pragma unroll
  for (int i = 0; i < 32; i++) fx[i] = xrow[i];
#pragma unroll
  for (int i = 0; i < 16; i++) fw[i] = wrow[i];
  cvt_store16(&As[0][aoff], fx);
  cvt_store16(&As[0][aoff + 16], fx + 16);
  cvt_store16(&Bs[0][boff], fw);
  __syncthreads();

  const int NKT = HID_ / 64;
  for (int kt = 0; kt < NKT; ++kt) {
    const int cur = kt & 1, nxt = cur ^ 1;
    const bool have_next = (kt + 1) < NKT;
    if (have_next) {
      const int k0 = (kt + 1) * 64;
#pragma unroll
      for (int i = 0; i < 32; i++) fx[i] = xrow[k0 + i];
#pragma unroll
      for (int i = 0; i < 16; i++) fw[i] = wrow[k0 + i];
    }
#pragma unroll
    for (int ks = 0; ks < 2; ks++) {
      v16bf a = load_afrag(As[cur], 16 * w, 72, 32 * ks, lane);
#pragma unroll
      for (int tn = 0; tn < 4; tn++) {
        v16bf b = load_bfrag(Bs[cur], 16 * tn, 72, 32 * ks, lane);
        acc[tn] = wmma_bf16(a, b, acc[tn]);
      }
    }
    if (have_next) {
      cvt_store16(&As[nxt][aoff], fx);
      cvt_store16(&As[nxt][aoff + 16], fx + 16);
      cvt_store16(&Bs[nxt][boff], fw);
    }
    __syncthreads();
  }

  // epilogue: bias + scatter to [B, NH, S, DH] bf16
  const int hl = lane >> 4, nl = lane & 15;
#pragma unroll
  for (int j = 0; j < 8; j++) {
    const int m = 16 * w + j + 8 * hl;
    const int gm = bm * 128 + m;
    const int bb = gm >> 11;          // / S_
    const int ss = gm & (S_ - 1);
#pragma unroll
    for (int tn = 0; tn < 4; tn++) {
      const int gn = bn * 64 + 16 * tn + nl;
      const float v = acc[tn][j] + bsel[gn];
      const int hh = gn >> 7, d = gn & (DH_ - 1);
      Osel[((size_t)(bb * NH_ + hh) * S_ + ss) * DH_ + d] = (bf16)v;
    }
  }
}

// ================= Kernel 2: RoPE (in place, bf16) ===========================
__global__ __launch_bounds__(256) void rope_kernel(bf16* __restrict__ Qb,
                                                   bf16* __restrict__ Kb) {
  const int idx = blockIdx.x * 256 + threadIdx.x;  // over B*NH*S*64 pairs
  bf16* base = (blockIdx.y == 0) ? Qb : Kb;
  const float scale = (blockIdx.y == 0) ? 0.08838834764831845f : 1.0f;  // DH^-0.5 on Q
  const int d  = idx & 63;
  const int s  = (idx >> 6) & (S_ - 1);
  const int bh = idx >> 17;  // / (S_*64)
  bf16* p = base + ((size_t)bh * S_ + s) * DH_;
  const float q0 = (float)p[d];
  const float q1 = (float)p[d + 64];
  const float freq = powf(10000.0f, -(float)(2 * d) / 128.0f);
  const float ang = (float)s * freq;
  const float c = cosf(ang), sn = sinf(ang);
  p[d]      = (bf16)((q0 * c - q1 * sn) * scale);
  p[d + 64] = (bf16)((q1 * c + q0 * sn) * scale);
}

// ================= Kernel 3: causal flash attention ==========================
// block = 32 queries of one (b,h); keys streamed in 64-wide tiles
__global__ __launch_bounds__(256) void attn_kernel(const bf16* __restrict__ Q,
                                                   const bf16* __restrict__ K,
                                                   const bf16* __restrict__ V,
                                                   bf16* __restrict__ ctx) {
  __shared__ bf16  Qs[32 * 136];
  __shared__ bf16  Ks[64 * 136];
  __shared__ bf16  Vts[128 * 72];   // V transposed: [d][key]
  __shared__ float Ss[32 * 68];
  __shared__ bf16  Ps[32 * 72];
  __shared__ float mrow[32], lrow[32], crow[32];

  const int t = threadIdx.x, lane = t & 31, w = t >> 5;
  const int mw = w >> 2;            // 0..1 : 16-row strip
  const int cw = w & 3;             // 0..3 : 16-col (score) / 32-col (O) group
  const int qi = blockIdx.x, bh = blockIdx.y;
  const int q0 = qi * 32;
  const bf16* Qb = Q + (size_t)bh * S_ * DH_;
  const bf16* Kb = K + (size_t)bh * S_ * DH_;
  const bf16* Vb = V + (size_t)bh * S_ * DH_;

  {  // stage Q tile 32x128
    const int r = t >> 3, c = (t & 7) * 16;
    const bf16* src = Qb + (size_t)(q0 + r) * DH_ + c;
    bf16* dst = Qs + r * 136 + c;
    *(bf16x8*)dst       = *(const bf16x8*)src;
    *(bf16x8*)(dst + 8) = *(const bf16x8*)(src + 8);
  }
  if (t < 32) { mrow[t] = -3.0e38f; lrow[t] = 0.f; }
  __syncthreads();

  v16bf qf[4];
#pragma unroll
  for (int ks = 0; ks < 4; ks++) qf[ks] = load_afrag(Qs, 16 * mw, 136, 32 * ks, lane);

  v8f o0 = {0.f, 0.f, 0.f, 0.f, 0.f, 0.f, 0.f, 0.f};
  v8f o1 = o0;

  const int nkb = (q0 >> 6) + 1;
  for (int kb = 0; kb < nkb; kb++) {
    const int k0 = kb * 64;
    {  // stage K tile 64x128 row-major
      const int r = t >> 2, c = (t & 3) * 32;
      const bf16* src = Kb + (size_t)(k0 + r) * DH_ + c;
      bf16* dst = Ks + r * 136 + c;
#pragma unroll
      for (int i = 0; i < 4; i++)
        *(bf16x8*)(dst + 8 * i) = *(const bf16x8*)(src + 8 * i);
    }
    {  // stage V tile 64x128 transposed -> Vts[d][key]
      const int key = t & 63, c = (t >> 6) * 32;
      const bf16* src = Vb + (size_t)(k0 + key) * DH_ + c;
#pragma unroll
      for (int i = 0; i < 32; i += 8) {
        bf16x8 vv = *(const bf16x8*)(src + i);
#pragma unroll
        for (int jj = 0; jj < 8; jj++) Vts[(c + i + jj) * 72 + key] = vv[jj];
      }
    }
    __syncthreads();

    // scores: each wave one 16x16 tile, contraction over DH=128 (4 WMMA)
    v8f sacc = {0.f, 0.f, 0.f, 0.f, 0.f, 0.f, 0.f, 0.f};
#pragma unroll
    for (int ks = 0; ks < 4; ks++) {
      v16bf bk = load_bfrag(Ks, 16 * cw, 136, 32 * ks, lane);
      sacc = wmma_bf16(qf[ks], bk, sacc);
    }
    {
      const int hl = lane >> 4, nl = lane & 15;
      const int n = 16 * cw + nl;
      const bool last = (kb == nkb - 1);
#pragma unroll
      for (int j = 0; j < 8; j++) {
        const int m = 16 * mw + j + 8 * hl;
        float v = sacc[j];
        if (last && (k0 + n > q0 + m)) v = -1.0e9f;  // causal mask, matches reference
        Ss[m * 68 + n] = v;
      }
    }
    __syncthreads();

    {  // online softmax: 8 threads per row, 8 cols each
      const int row = t >> 3, p = t & 7;
      const float* srow = Ss + row * 68 + p * 8;
      float vbuf[8];
      float mx = -3.0e38f;
#pragma unroll
      for (int i = 0; i < 8; i++) { vbuf[i] = srow[i]; mx = fmaxf(mx, vbuf[i]); }
      mx = fmaxf(mx, __shfl_xor(mx, 1));
      mx = fmaxf(mx, __shfl_xor(mx, 2));
      mx = fmaxf(mx, __shfl_xor(mx, 4));
      const float mold = mrow[row];
      const float mnew = fmaxf(mold, mx);
      const float corr = __expf(mold - mnew);
      float sum = 0.f;
      bf16* prow = Ps + row * 72 + p * 8;
#pragma unroll
      for (int i = 0; i < 8; i++) {
        const float e = __expf(vbuf[i] - mnew);
        sum += e;
        prow[i] = (bf16)e;
      }
      sum += __shfl_xor(sum, 1);
      sum += __shfl_xor(sum, 2);
      sum += __shfl_xor(sum, 4);
      if (p == 0) {
        crow[row] = corr;
        mrow[row] = mnew;
        lrow[row] = lrow[row] * corr + sum;
      }
    }
    __syncthreads();

    {  // rescale accumulators
      const int hl = lane >> 4;
#pragma unroll
      for (int j = 0; j < 8; j++) {
        const float c = crow[16 * mw + j + 8 * hl];
        o0[j] *= c;
        o1[j] *= c;
      }
    }
    // P(32x64 bf16) x V(64x128): each wave 2 tiles over d = 32*cw + {0,16}
#pragma unroll
    for (int ks = 0; ks < 2; ks++) {
      v16bf pa  = load_afrag(Ps, 16 * mw, 72, 32 * ks, lane);
      v16bf bv0 = load_bfrag(Vts, 32 * cw, 72, 32 * ks, lane);
      v16bf bv1 = load_bfrag(Vts, 32 * cw + 16, 72, 32 * ks, lane);
      o0 = wmma_bf16(pa, bv0, o0);
      o1 = wmma_bf16(pa, bv1, o1);
    }
    __syncthreads();
  }

  {  // epilogue: normalize and scatter to [B, S, NH, DH] bf16
    const int hl = lane >> 4, nl = lane & 15;
    const int bb = bh >> 4, hh = bh & (NH_ - 1);
#pragma unroll
    for (int j = 0; j < 8; j++) {
      const int m = 16 * mw + j + 8 * hl;
      const float inv = 1.0f / lrow[m];
      const int srows = q0 + m;
#pragma unroll
      for (int tn = 0; tn < 2; tn++) {
        const int d = 32 * cw + 16 * tn + nl;
        ctx[(((size_t)(bb * S_ + srows)) * NH_ + hh) * DH_ + d] =
            (bf16)(((tn == 0) ? o0[j] : o1[j]) * inv);
      }
    }
  }
}

// ================= Kernel 4: output projection (bf16 A, f32 out) =============
// Same 128x64 double-buffered scheme; A is already bf16.
__global__ __launch_bounds__(256) void out_gemm_kernel(
    const bf16* __restrict__ A, const float* __restrict__ Wo,
    const float* __restrict__ bo, float* __restrict__ Y) {
  __shared__ bf16 As[2][128 * 72];
  __shared__ bf16 Bs[2][64 * 72];
  const int t = threadIdx.x, lane = t & 31, w = t >> 5;
  const int bm = blockIdx.x, bn = blockIdx.y;

  v8f acc[4];
#pragma unroll
  for (int i = 0; i < 4; i++) acc[i] = (v8f){0.f, 0.f, 0.f, 0.f, 0.f, 0.f, 0.f, 0.f};

  const int ar = t >> 1, ac = (t & 1) * 32;
  const int br = t >> 2, bc = (t & 3) * 16;
  const bf16* arow = A + (size_t)(bm * 128 + ar) * HID_ + ac;
  const float* wrow = Wo + (size_t)(bn * 64 + br) * HID_ + bc;
  const int aoff = ar * 72 + ac;
  const int boff = br * 72 + bc;

  bf16x8 ha[4];
  float fw[16];
#pragma unroll
  for (int i = 0; i < 4; i++) ha[i] = *(const bf16x8*)(arow + 8 * i);
#pragma unroll
  for (int i = 0; i < 16; i++) fw[i] = wrow[i];
#pragma unroll
  for (int i = 0; i < 4; i++) *(bf16x8*)(&As[0][aoff + 8 * i]) = ha[i];
  cvt_store16(&Bs[0][boff], fw);
  __syncthreads();

  const int NKT = HID_ / 64;
  for (int kt = 0; kt < NKT; ++kt) {
    const int cur = kt & 1, nxt = cur ^ 1;
    const bool have_next = (kt + 1) < NKT;
    if (have_next) {
      const int k0 = (kt + 1) * 64;
#pragma unroll
      for (int i = 0; i < 4; i++) ha[i] = *(const bf16x8*)(arow + k0 + 8 * i);
#pragma unroll
      for (int i = 0; i < 16; i++) fw[i] = wrow[k0 + i];
    }
#pragma unroll
    for (int ks = 0; ks < 2; ks++) {
      v16bf a = load_afrag(As[cur], 16 * w, 72, 32 * ks, lane);
#pragma unroll
      for (int tn = 0; tn < 4; tn++) {
        v16bf b = load_bfrag(Bs[cur], 16 * tn, 72, 32 * ks, lane);
        acc[tn] = wmma_bf16(a, b, acc[tn]);
      }
    }
    if (have_next) {
#pragma unroll
      for (int i = 0; i < 4; i++) *(bf16x8*)(&As[nxt][aoff + 8 * i]) = ha[i];
      cvt_store16(&Bs[nxt][boff], fw);
    }
    __syncthreads();
  }

  const int hl = lane >> 4, nl = lane & 15;
#pragma unroll
  for (int j = 0; j < 8; j++) {
    const int gm = bm * 128 + 16 * w + j + 8 * hl;
#pragma unroll
    for (int tn = 0; tn < 4; tn++) {
      const int gn = bn * 64 + 16 * tn + nl;
      Y[(size_t)gm * HID_ + gn] = acc[tn][j] + bo[gn];
    }
  }
}

// ================= Host launcher ============================================
extern "C" void kernel_launch(void* const* d_in, const int* in_sizes, int n_in,
                              void* d_out, int out_size, void* d_ws,
                              size_t ws_size, hipStream_t stream) {
  (void)in_sizes; (void)n_in; (void)out_size; (void)ws_size;
  const float* X  = (const float*)d_in[0];
  // d_in[1] = attention_mask (pure causal; implemented analytically)
  const float* Wq = (const float*)d_in[2];
  const float* bq = (const float*)d_in[3];
  const float* Wk = (const float*)d_in[4];
  const float* bk = (const float*)d_in[5];
  const float* Wv = (const float*)d_in[6];
  const float* bv = (const float*)d_in[7];
  const float* Wo = (const float*)d_in[8];
  const float* bo = (const float*)d_in[9];
  float* out = (float*)d_out;

  const size_t NE = (size_t)B_ * NH_ * S_ * DH_;  // 8,388,608 elems per tensor
  bf16* qbuf = (bf16*)d_ws;
  bf16* kbuf = qbuf + NE;
  bf16* vbuf = kbuf + NE;
  bf16* cbuf = vbuf + NE;   // total ws use: 4*NE*2 = 64 MiB

  dim3 blk(256);
  qkv_gemm_kernel<<<dim3(MTOT / 128, HID_ / 64, 3), blk, 0, stream>>>(
      X, Wq, Wk, Wv, bq, bk, bv, qbuf, kbuf, vbuf);
  rope_kernel<<<dim3((unsigned)((NE / 2) / 256), 2), blk, 0, stream>>>(qbuf, kbuf);
  attn_kernel<<<dim3(S_ / 32, B_ * NH_), blk, 0, stream>>>(qbuf, kbuf, vbuf, cbuf);
  out_gemm_kernel<<<dim3(MTOT / 128, HID_ / 64), blk, 0, stream>>>(cbuf, Wo, bo, out);
}
